// DTransformerLayer_27917287424233
// MI455X (gfx1250) — compile-verified, matrix operand order checked by reading, not run on hardware
//
#include <hip/hip_runtime.h>

// ---------------------------------------------------------------------------
// DTransformer layer for MI455X (gfx1250): bf16 WMMA GEMMs (async-LDS staged,
// double buffered) + WMMA attention, f32 softmax/decay/LayerNorm. wave32.
// ---------------------------------------------------------------------------

#define Bn 4
#define Sn 1024
#define Dn 1024
#define Hn 16
#define DKn 64
#define NEGV (-1.0e32f)

typedef __attribute__((ext_vector_type(16))) __bf16 v16bf;
typedef __attribute__((ext_vector_type(8)))  float  v8f;

// ---- WMMA fragment loaders (ISA 7.12.2 layouts, wave32) -------------------
// A (16x32 bf16, row-major, stride ld): lane&15 = M row; half-wave picks K set.
// Also serves as the B operand when the source is B^T stored row-major (NxK).
__device__ __forceinline__ v16bf frag_a_rowmajor(const __bf16* p, int ld) {
  int lane = threadIdx.x & 31;
  int m    = lane & 15;
  int kb   = (lane >> 4) << 3;            // 0 or 8
  const __bf16* row = p + (size_t)m * ld;
  v16bf a;
#pragma unroll
  for (int v = 0; v < 8; ++v) {
    int k0 = (v < 4) ? (kb + 2 * v) : (16 + kb + 2 * (v - 4));
    a[2 * v]     = row[k0];
    a[2 * v + 1] = row[k0 + 1];
  }
  return a;
}

// Same layout but source is f32 in LDS (convert to bf16 on the fly).
__device__ __forceinline__ v16bf frag_a_lds_f32(const float* p, int ld) {
  int lane = threadIdx.x & 31;
  int m    = lane & 15;
  int kb   = (lane >> 4) << 3;
  const float* row = p + (size_t)m * ld;
  v16bf a;
#pragma unroll
  for (int v = 0; v < 8; ++v) {
    int k0 = (v < 4) ? (kb + 2 * v) : (16 + kb + 2 * (v - 4));
    a[2 * v]     = (__bf16)row[k0];
    a[2 * v + 1] = (__bf16)row[k0 + 1];
  }
  return a;
}

__device__ __forceinline__ v8f wmma_bf16(v16bf a, v16bf b, v8f c) {
  return __builtin_amdgcn_wmma_f32_16x16x32_bf16(false, a, false, b,
                                                 (short)0, c, false, false);
}

// ---- CDNA5 async copy helpers (ASYNCcnt path) -----------------------------
// global -> LDS, 16B per issue, no VGPR return data; tracked with ASYNCcnt.
__device__ __forceinline__ void async_b128(unsigned lds_off,
                                           const __bf16* gaddr) {
  asm volatile("global_load_async_to_lds_b128 %0, %1, off"
               :
               : "v"(lds_off), "v"((unsigned long long)(uintptr_t)gaddr)
               : "memory");
}
__device__ __forceinline__ void wait_async0() {
  asm volatile("s_wait_asynccnt 0" ::: "memory");
}

// ---- elementwise converts -------------------------------------------------
__global__ void cvt_f32_bf16(const float* __restrict__ src,
                             __bf16* __restrict__ dst, int n) {
  int i = blockIdx.x * blockDim.x + threadIdx.x;
  if (i < n) dst[i] = (__bf16)src[i];
}

// WT[n*dim + k] = W[k*dim + n]  (bf16 transposed weights for B operands)
__global__ void transpose_cvt(const float* __restrict__ W,
                              __bf16* __restrict__ WT, int dim) {
  int i = blockIdx.x * blockDim.x + threadIdx.x;
  if (i < dim * dim) {
    int n = i / dim, k = i - n * dim;
    WT[i] = (__bf16)W[(size_t)k * dim + n];
  }
}

// vT[((b*H + h)*DK + dk)*S + s] = vh[(b*S + s)*D + h*DK + dk]
// Gives the P*V B-operand a row-major (N=dk, K=s) layout -> contiguous b128s.
__global__ void transpose_v(const __bf16* __restrict__ vh,
                            __bf16* __restrict__ vT) {
  int i = blockIdx.x * blockDim.x + threadIdx.x;   // over B*H*DK*S
  int s  = i & (Sn - 1);
  int t  = i >> 10;
  int dk = t & (DKn - 1);
  int bh = t >> 6;                                  // b*H + h
  int b  = bh >> 4;
  int h  = bh & (Hn - 1);
  vT[i] = vh[((size_t)(b * Sn + s)) * Dn + h * DKn + dk];
}

// ---- WMMA GEMM: C[M,N] = A[M,K] * BT[N,K]^T + bias ------------------------
// 128x128 block tile, K in 32-chunks double-buffered in LDS via async loads.
// block = 256 threads (8 waves); wave w -> (mrow = w&3)*32 x (ncol = w>>2)*64
// sub-tile = 2x4 WMMA accumulators.
#define BKc 32
__global__ __launch_bounds__(256)
void gemm_bf16(const __bf16* __restrict__ A, const __bf16* __restrict__ BT,
               const float* __restrict__ bias,
               __bf16* __restrict__ Cb, float* __restrict__ Cf,
               int M, int N, int K) {
  __shared__ __align__(16) __bf16 Abuf[2][128 * BKc];   // 2 x 8 KB
  __shared__ __align__(16) __bf16 Bbuf[2][128 * BKc];   // 2 x 8 KB

  const int tid  = threadIdx.x;
  const int wave = tid >> 5;
  const int lane = tid & 31;
  const int mrow = wave & 3;         // 4 M-groups of 32 rows
  const int ncol = wave >> 2;        // 2 N-groups of 64 cols
  const int m0   = blockIdx.x * 128;
  const int n0   = blockIdx.y * 128;

  // each thread fills 2 x 16B of A and 2 x 16B of B per K-chunk
  // segment = 16B = 8 bf16; a 32-wide row = 4 segments; 128 rows = 512 segs
  const int seg0 = tid;              // + 256 for the second one
  const int nk   = K / BKc;

  auto issue_chunk = [&](int kc, int buf) {
#pragma unroll
    for (int i = 0; i < 2; ++i) {
      int seg  = seg0 + i * 256;
      int row  = seg >> 2;
      int part = seg & 3;
      unsigned loff = (unsigned)(uintptr_t)(&Abuf[buf][row * BKc + part * 8]);
      const __bf16* g = A + (size_t)(m0 + row) * K + kc * BKc + part * 8;
      async_b128(loff, g);
      unsigned loffb = (unsigned)(uintptr_t)(&Bbuf[buf][row * BKc + part * 8]);
      const __bf16* gb = BT + (size_t)(n0 + row) * K + kc * BKc + part * 8;
      async_b128(loffb, gb);
    }
  };

  v8f acc[2][4];
#pragma unroll
  for (int mt = 0; mt < 2; ++mt)
#pragma unroll
    for (int nt = 0; nt < 4; ++nt) acc[mt][nt] = {};

  issue_chunk(0, 0);

  for (int kc = 0; kc < nk; ++kc) {
    int cur = kc & 1;
    wait_async0();
    __syncthreads();                  // fills landed; buf[cur^1] free to refill
    if (kc + 1 < nk) issue_chunk(kc + 1, cur ^ 1);

    v16bf af[2], bf[4];
#pragma unroll
    for (int mt = 0; mt < 2; ++mt)
      af[mt] = frag_a_rowmajor(&Abuf[cur][(mrow * 32 + mt * 16) * BKc], BKc);
#pragma unroll
    for (int nt = 0; nt < 4; ++nt)
      bf[nt] = frag_a_rowmajor(&Bbuf[cur][(ncol * 64 + nt * 16) * BKc], BKc);
#pragma unroll
    for (int mt = 0; mt < 2; ++mt)
#pragma unroll
      for (int nt = 0; nt < 4; ++nt)
        acc[mt][nt] = wmma_bf16(af[mt], bf[nt], acc[mt][nt]);
    __syncthreads();                  // done reading buf[cur] before next refill
  }

  // store: C layout -> lane&15 = N, half-wave = M>=8, VGPR r = M&7
  const int nl0 = n0 + ncol * 64;
  const int mb0 = m0 + mrow * 32 + ((lane >> 4) << 3);
#pragma unroll
  for (int mt = 0; mt < 2; ++mt) {
#pragma unroll
    for (int nt = 0; nt < 4; ++nt) {
      int nl = nl0 + nt * 16 + (lane & 15);
      int mb = mb0 + mt * 16;
      float bv = bias[nl];
#pragma unroll
      for (int r = 0; r < 8; ++r) {
        int m = mb + r;
        float v = acc[mt][nt][r] + bv;
        if (Cb) Cb[(size_t)m * N + nl] = (__bf16)v;
        if (Cf) Cf[(size_t)m * N + nl] = v;
      }
    }
  }
}

// ---- attention ------------------------------------------------------------
// grid = B*H*(S/16), block = 128 (4 waves). 64 KB LDS score strip.
// vT is the per-head transposed V: [B,H,DK,S] bf16.
template <bool WINDOW, bool MAXOUT>
__global__ __launch_bounds__(128)
void attn_kernel(const __bf16* __restrict__ qh, const __bf16* __restrict__ kh,
                 const __bf16* __restrict__ vT, const float* __restrict__ gammas,
                 __bf16* __restrict__ ao) {
  __shared__ float sc[16 * Sn];           // exactly 64 KB

  int qt = blockIdx.x & (Sn / 16 - 1);
  int bh = blockIdx.x >> 6;
  int b  = bh >> 4;
  int h  = bh & (Hn - 1);
  int q0 = qt * 16;
  int tid  = threadIdx.x;
  int wave = tid >> 5;
  int lane = tid & 31;

  // --- phase 1: scores[16 x S] = Q K^T / sqrt(DK) via WMMA ---
  const __bf16* qbase = qh + ((size_t)(b * Sn + q0)) * Dn + h * DKn;
  for (int i = 0; i < 16; ++i) {
    int kt = wave + 4 * i;                 // 64 key tiles split over 4 waves
    const __bf16* kbase = kh + ((size_t)(b * Sn + kt * 16)) * Dn + h * DKn;
    v8f cc = {};
#pragma unroll
    for (int kk = 0; kk < DKn; kk += 32) {
      v16bf a  = frag_a_rowmajor(qbase + kk, Dn);
      v16bf bt = frag_a_rowmajor(kbase + kk, Dn);  // keys row-major == B^T
      cc = wmma_bf16(a, bt, cc);
    }
    int nl = lane & 15;
    int mb = (lane >> 4) << 3;
#pragma unroll
    for (int r = 0; r < 8; ++r)
      sc[(mb + r) * Sn + kt * 16 + nl] = cc[r] * 0.125f;  // 1/sqrt(64)
  }
  __syncthreads();

  // --- phase 2: softmax -> distance decay -> softmax (-> maxout), f32 ---
  // one query row per 8 consecutive lanes; shuffle-width-8 reductions/scans
  {
    int row = tid >> 3;
    int seg = tid & 7;
    int c0  = seg * 128;
    int qg  = q0 + row;
    float* srow = sc + row * Sn;
    float g = -fabsf(gammas[h]);

    float m1 = NEGV;
    for (int j = 0; j < 128; ++j) {
      int c = c0 + j;
      bool valid = WINDOW ? (c < qg && c >= qg - 19) : (c < qg);
      float s = valid ? srow[c] : NEGV;
      m1 = fmaxf(m1, s);
    }
    for (int d = 1; d < 8; d <<= 1) m1 = fmaxf(m1, __shfl_xor(m1, d, 8));

    float part = 0.f;
    for (int j = 0; j < 128; ++j) {
      int c = c0 + j;
      bool valid = WINDOW ? (c < qg && c >= qg - 19) : (c < qg);
      float s = valid ? srow[c] : NEGV;
      part += __expf(s - m1);
    }
    float tot = part;
    for (int d = 1; d < 8; d <<= 1) tot += __shfl_xor(tot, d, 8);
    float pre = part;
    for (int d = 1; d < 8; d <<= 1) {
      float t = __shfl_up(pre, (unsigned)d, 8);
      if (seg >= d) pre += t;
    }
    float run = pre - part;                 // exclusive prefix of exp-sums
    float inv_tot = 1.0f / tot;

    for (int j = 0; j < 128; ++j) {         // cumsum + decay, write s2
      int c = c0 + j;
      bool valid = WINDOW ? (c < qg && c >= qg - 19) : (c < qg);
      float s = valid ? srow[c] : NEGV;
      float e = __expf(s - m1);
      run += e;
      float distcum = run * inv_tot;        // inclusive cumsum of p
      float pe = fabsf((float)(c - qg));    // |k - q|
      float dd = sqrtf(fmaxf((1.0f - distcum) * pe, 0.0f));
      float te = __expf(dd * g);
      te = fminf(fmaxf(te, 1e-5f), 1e5f);
      srow[c] = valid ? s * te : NEGV;
    }

    float m2 = NEGV;
    for (int j = 0; j < 128; ++j) m2 = fmaxf(m2, srow[c0 + j]);
    for (int d = 1; d < 8; d <<= 1) m2 = fmaxf(m2, __shfl_xor(m2, d, 8));
    float p2 = 0.f;
    for (int j = 0; j < 128; ++j) p2 += __expf(srow[c0 + j] - m2);
    float tot2 = p2;
    for (int d = 1; d < 8; d <<= 1) tot2 += __shfl_xor(tot2, d, 8);
    float inv2 = 1.0f / tot2;

    float pmax = 0.f;
    for (int j = 0; j < 128; ++j) {
      int c = c0 + j;
      bool valid = WINDOW ? (c < qg && c >= qg - 19) : (c < qg);
      float p = valid ? __expf(srow[c] - m2) * inv2 : 0.f;
      srow[c] = p;
      pmax = fmaxf(pmax, p);
    }
    if (MAXOUT) {
      for (int d = 1; d < 8; d <<= 1) pmax = fmaxf(pmax, __shfl_xor(pmax, d, 8));
      float scale = fminf(1.0f / pmax, 5.0f);
      for (int j = 0; j < 128; ++j) srow[c0 + j] *= scale;
    }
  }
  __syncthreads();

  // --- phase 3: O[16 x 64] = P * V via WMMA; wave w owns dk-tile w ---
  // B operand from vT rows (N=dk, K=s): two 16B runs (32B) per lane, b128-merged.
  {
    v8f acc = {};
    const __bf16* vtbase = vT + ((size_t)bh * DKn + wave * 16) * Sn;
    for (int kc = 0; kc < Sn / 32; ++kc) {
      v16bf a  = frag_a_lds_f32(sc + kc * 32, Sn);
      v16bf bb = frag_a_rowmajor(vtbase + kc * 32, Sn);
      acc = wmma_bf16(a, bb, acc);
    }
    int nl = lane & 15;
    int mb = (lane >> 4) << 3;
#pragma unroll
    for (int r = 0; r < 8; ++r) {
      int m = mb + r;
      ao[((size_t)(b * Sn + q0 + m)) * Dn + h * DKn + wave * 16 + nl] =
          (__bf16)acc[r];
    }
  }
}

// ---- residual + dual LayerNorm -------------------------------------------
__device__ float block_sum_256(float v) {
  __shared__ float buf[9];
  for (int d = 16; d > 0; d >>= 1) v += __shfl_xor(v, d, 32);
  int lane = threadIdx.x & 31, w = threadIdx.x >> 5;
  __syncthreads();
  if (lane == 0) buf[w] = v;
  __syncthreads();
  if (threadIdx.x == 0) {
    float t = 0.f;
    for (int i = 0; i < 8; ++i) t += buf[i];
    buf[8] = t;
  }
  __syncthreads();
  return buf[8];
}

__global__ __launch_bounds__(256)
void residual_ln(const float* __restrict__ query, const float* __restrict__ outf,
                 const float* __restrict__ outwf, const float* __restrict__ lnw,
                 const float* __restrict__ lnb, float* __restrict__ o0,
                 float* __restrict__ o1) {
  size_t row = blockIdx.x;
  int tid = threadIdx.x;
  const float* q  = query + row * Dn;
  const float* of = outf  + row * Dn;
  const float* ow = outwf + row * Dn;

  float x1[4], x2[4];
  float s = 0.f, ss = 0.f;
#pragma unroll
  for (int i = 0; i < 4; ++i) {
    int c = tid + i * 256;
    x1[i] = q[c] + of[c];
    s += x1[i];
    ss += x1[i] * x1[i];
  }
  float mu  = block_sum_256(s) * (1.0f / Dn);
  float ex2 = block_sum_256(ss) * (1.0f / Dn);
  float inv = rsqrtf(ex2 - mu * mu + 1e-5f);
#pragma unroll
  for (int i = 0; i < 4; ++i) {
    int c = tid + i * 256;
    o0[row * Dn + c] = (x1[i] - mu) * inv * lnw[c] + lnb[c];
  }

  s = 0.f; ss = 0.f;
#pragma unroll
  for (int i = 0; i < 4; ++i) {
    int c = tid + i * 256;
    x2[i] = x1[i] + ow[c];
    s += x2[i];
    ss += x2[i] * x2[i];
  }
  mu  = block_sum_256(s) * (1.0f / Dn);
  ex2 = block_sum_256(ss) * (1.0f / Dn);
  inv = rsqrtf(ex2 - mu * mu + 1e-5f);
#pragma unroll
  for (int i = 0; i < 4; ++i) {
    int c = tid + i * 256;
    o1[row * Dn + c] = (x2[i] - mu) * inv * lnw[c] + lnb[c];
  }
}

// ---------------------------------------------------------------------------
extern "C" void kernel_launch(void* const* d_in, const int* in_sizes, int n_in,
                              void* d_out, int out_size, void* d_ws,
                              size_t ws_size, hipStream_t stream) {
  const float* query  = (const float*)d_in[0];
  const float* key    = (const float*)d_in[1];
  const float* values = (const float*)d_in[2];
  // d_in[3] = lens : unused by the reference computation
  const float* Wq   = (const float*)d_in[4];
  const float* bq   = (const float*)d_in[5];
  const float* Wqw  = (const float*)d_in[6];
  const float* bqw  = (const float*)d_in[7];
  const float* Wv   = (const float*)d_in[8];
  const float* bv   = (const float*)d_in[9];
  const float* Wo   = (const float*)d_in[10];
  const float* bo   = (const float*)d_in[11];
  const float* Wow  = (const float*)d_in[12];
  const float* bow  = (const float*)d_in[13];
  const float* gam  = (const float*)d_in[14];
  const float* lnw  = (const float*)d_in[15];
  const float* lnb  = (const float*)d_in[16];

  const size_t NBS = (size_t)Bn * Sn * Dn;   // 4 Mi elems
  const size_t NW  = (size_t)Dn * Dn;        // 1 Mi elems

  char* p = (char*)d_ws;
  __bf16* qx   = (__bf16*)p; p += NBS * 2;
  __bf16* kx   = (__bf16*)p; p += NBS * 2;
  __bf16* vx   = (__bf16*)p; p += NBS * 2;
  __bf16* WqT  = (__bf16*)p; p += NW * 2;
  __bf16* WqwT = (__bf16*)p; p += NW * 2;
  __bf16* WvT  = (__bf16*)p; p += NW * 2;
  __bf16* WoT  = (__bf16*)p; p += NW * 2;
  __bf16* WowT = (__bf16*)p; p += NW * 2;
  __bf16* qhB  = (__bf16*)p; p += NBS * 2;
  __bf16* khB  = (__bf16*)p; p += NBS * 2;
  __bf16* vhB  = (__bf16*)p; p += NBS * 2;
  __bf16* vTB  = (__bf16*)p; p += NBS * 2;
  __bf16* qwB  = (__bf16*)p; p += NBS * 2;
  __bf16* kwB  = (__bf16*)p; p += NBS * 2;
  __bf16* aoB  = (__bf16*)p; p += NBS * 2;
  __bf16* aowB = (__bf16*)p; p += NBS * 2;
  float*  outF  = (float*)p; p += NBS * 4;
  float*  outwF = (float*)p; p += NBS * 4;

  float* o0 = (float*)d_out;
  float* o1 = o0 + NBS;

  // 1) converts
  cvt_f32_bf16<<<(int)(NBS / 256), 256, 0, stream>>>(query,  qx, (int)NBS);
  cvt_f32_bf16<<<(int)(NBS / 256), 256, 0, stream>>>(key,    kx, (int)NBS);
  cvt_f32_bf16<<<(int)(NBS / 256), 256, 0, stream>>>(values, vx, (int)NBS);
  transpose_cvt<<<(int)(NW / 256), 256, 0, stream>>>(Wq,  WqT,  Dn);
  transpose_cvt<<<(int)(NW / 256), 256, 0, stream>>>(Wqw, WqwT, Dn);
  transpose_cvt<<<(int)(NW / 256), 256, 0, stream>>>(Wv,  WvT,  Dn);
  transpose_cvt<<<(int)(NW / 256), 256, 0, stream>>>(Wo,  WoT,  Dn);
  transpose_cvt<<<(int)(NW / 256), 256, 0, stream>>>(Wow, WowT, Dn);

  // 2) input projections (kq_same: q_ and k_ share Wq; qw/kw share Wqw)
  const int M = Bn * Sn;                    // 4096
  dim3 gg(M / 128, Dn / 128);               // 32 x 8 blocks of 128x128
  gemm_bf16<<<gg, 256, 0, stream>>>(qx, WqT,  bq,  qhB, nullptr, M, Dn, Dn);
  gemm_bf16<<<gg, 256, 0, stream>>>(kx, WqT,  bq,  khB, nullptr, M, Dn, Dn);
  gemm_bf16<<<gg, 256, 0, stream>>>(vx, WvT,  bv,  vhB, nullptr, M, Dn, Dn);
  gemm_bf16<<<gg, 256, 0, stream>>>(qx, WqwT, bqw, qwB, nullptr, M, Dn, Dn);
  gemm_bf16<<<gg, 256, 0, stream>>>(kx, WqwT, bqw, kwB, nullptr, M, Dn, Dn);

  // 2b) per-head transpose of V for coalesced P*V B-operands
  transpose_v<<<(int)(NBS / 256), 256, 0, stream>>>(vhB, vTB);

  // 3) attentions: strict-causal + maxout; band[-19,-1] without maxout
  int agrid = Bn * Hn * (Sn / 16);          // 4096
  attn_kernel<false, true ><<<agrid, 128, 0, stream>>>(qhB, khB, vTB, gam, aoB);
  attn_kernel<true,  false><<<agrid, 128, 0, stream>>>(qwB, kwB, vTB, gam, aowB);

  // 4) output projections (f32 results for residual+LN)
  gemm_bf16<<<gg, 256, 0, stream>>>(aoB,  WoT,  bo,  nullptr, outF,  M, Dn, Dn);
  gemm_bf16<<<gg, 256, 0, stream>>>(aowB, WowT, bow, nullptr, outwF, M, Dn, Dn);

  // 5) residual + dual LayerNorm -> both tuple outputs
  residual_ln<<<M, 256, 0, stream>>>(query, outF, outwF, lnw, lnb, o0, o1);
}
